// CrossAttention_13589276524754
// MI455X (gfx1250) — matrix operand order repaired and checked by reading, not compiled, not run on hardware
//
#include <hip/hip_runtime.h>

typedef unsigned short u16;
typedef __attribute__((ext_vector_type(8)))  __bf16 bf16x8;
typedef __attribute__((ext_vector_type(16))) __bf16 v16bf;
typedef __attribute__((ext_vector_type(8)))  float  v8f;
typedef __attribute__((ext_vector_type(4)))  unsigned u32x4;
typedef __attribute__((ext_vector_type(8)))  int      i32x8;
typedef __attribute__((ext_vector_type(4)))  int      i32x4;

#if defined(__has_builtin)
#  if __has_builtin(__builtin_amdgcn_tensor_load_to_lds)
#    define HAVE_TDM 1
#  endif
#endif
#ifndef HAVE_TDM
#  define HAVE_TDM 0
#endif

// ---------- bf16 <-> f32 helpers (RNE), buffers held as raw u16 ----------
__device__ __forceinline__ u16 f2bf(float f) {
    unsigned u = __float_as_uint(f);
    unsigned r = u + 0x7FFFu + ((u >> 16) & 1u);
    return (u16)(r >> 16);
}
__device__ __forceinline__ float bf2f(u16 h) {
    return __uint_as_float(((unsigned)h) << 16);
}
__device__ __forceinline__ v16bf cat8(bf16x8 lo, bf16x8 hi) {
    return __builtin_shufflevector(lo, hi, 0,1,2,3,4,5,6,7,8,9,10,11,12,13,14,15);
}
__device__ __forceinline__ v8f wmma_bf16(v16bf a, v16bf b, v8f c) {
    return __builtin_amdgcn_wmma_f32_16x16x32_bf16(false, a, false, b, (short)0, c, false, false);
}

#if HAVE_TDM
// 2-D TDM tile load: 32 (contiguous K) x 64 rows of 2-byte elements,
// global (row stride `stride` elems) -> LDS (packed [64][32]).
// clang-23 toolchain: 6-arg builtin (g0 v4u, g1 v8i, g2 v4i, g3 v4i, ext v8i, cpol).
__device__ __forceinline__ void tdm_load_2d(unsigned lds_off, const u16* g, int stride)
{
    unsigned long long ga = (unsigned long long)(size_t)g;
    u32x4 g0;
    g0[0] = 1u;                                             // count=1, user mode
    g0[1] = lds_off;                                        // lds_addr (bytes)
    g0[2] = (unsigned)ga;                                   // global_addr[31:0]
    g0[3] = ((unsigned)(ga >> 32) & 0x01FFFFFFu) | (2u << 30); // addr[56:32] | type=2
    unsigned td0 = (unsigned)stride;                        // tensor_dim0 = row length
    unsigned td1 = 1u << 20;                                // rows: effectively unbounded
    unsigned long long st0 = (unsigned long long)(unsigned)stride;
    i32x8 g1;
    g1[0] = (int)(1u << 16);                                // data_size = 1 -> 2 bytes
    g1[1] = (int)((td0 & 0xFFFFu) << 16);                   // tensor_dim0[15:0]
    g1[2] = (int)(((td0 >> 16) & 0xFFFFu) | ((td1 & 0xFFFFu) << 16)); // dim0 hi | dim1 lo
    g1[3] = (int)(((td1 >> 16) & 0xFFFFu) | (32u << 16));   // dim1 hi | tile_dim0=32
    g1[4] = (int)64u;                                       // tile_dim1=64, tile_dim2=0
    g1[5] = (int)(unsigned)(st0 & 0xFFFFFFFFu);             // dim0_stride[31:0]
    g1[6] = (int)(unsigned)((st0 >> 32) & 0xFFFFu);         // dim0_stride[47:32]
    g1[7] = 0;                                              // dim1_stride (unused, 2D)
    i32x4 gz4 = {0, 0, 0, 0};
    i32x8 gz8 = {0, 0, 0, 0, 0, 0, 0, 0};
    __builtin_amdgcn_tensor_load_to_lds(g0, g1, gz4, gz4, gz8, 0);
}
#endif

// =====================================================================
// elementwise f32 -> bf16 (weights, layout preserved)
// =====================================================================
__global__ void f32_to_bf16_kernel(const float* __restrict__ X, u16* __restrict__ Y, int n) {
    int i = blockIdx.x * 256 + threadIdx.x;
    if (i < n) Y[i] = f2bf(X[i]);
}

// channel-major f32 [b][C][N] -> position-major bf16 [b][N][C]
__global__ void cm_to_pm_bf16_kernel(const float* __restrict__ X, u16* __restrict__ Y,
                                     int C, int Nn) {
    int idx = blockIdx.x * 256 + threadIdx.x;
    if (idx < C * Nn) {
        int n = idx / C, c = idx - n * C;   // consecutive threads -> consecutive c (coalesced writes)
        Y[((size_t)blockIdx.z * Nn + n) * C + c] =
            f2bf(X[((size_t)blockIdx.z * C + c) * Nn + n]);
    }
}

// =====================================================================
// GEMM: Y[b] = W[Cout,Cin] @ Xpm[b]^T + bias (+resid).
// X is position-major [B][Ncol][rowlen]; virtual K-concat of X0 (K0 wide)
// and X1 (Cin-K0 wide). Tiles staged via TDM (double-buffered) — both A
// and B tiles are K-contiguous, no transpose anywhere.
// Outputs: Yf (f32, channel-major), Ybf_cm (bf16 ch-major), Ypm (bf16 pos-major).
// =====================================================================
__global__ __launch_bounds__(128)
void gemm_bf16_kernel(const u16* __restrict__ W, const u16* __restrict__ X0,
                      const u16* __restrict__ X1, int K0,
                      const float* __restrict__ bias, const float* __restrict__ resid,
                      float* __restrict__ Yf, u16* __restrict__ Ybf_cm,
                      u16* __restrict__ Ypm,
                      int Cout, int Cin, int Ncol)
{
    __shared__ u16 smem[2 * 4096];   // 2 stages x (A[64][32] + B[64][32])

    const int b   = blockIdx.z;
    const int tM  = blockIdx.y * 64;
    const int tN  = blockIdx.x * 64;
    const int tid = threadIdx.x;
    const int lane = tid & 31, w = tid >> 5;
    const int wm = w >> 1, wn = w & 1;
    const int nl = lane & 15, half = lane >> 4;
    const int K1 = Cin - K0;

    v8f acc[2][2] = {};

    auto xrow = [&](int k0) -> const u16* {   // &X[b][tN][k] for virtual k
        if (k0 < K0) return X0 + ((size_t)b * Ncol + tN) * K0 + k0;
        return X1 + ((size_t)b * Ncol + tN) * K1 + (k0 - K0);
    };
    auto rowlen = [&](int k0) -> int { return (k0 < K0) ? K0 : K1; };

#if HAVE_TDM
    const unsigned ldsBase = (unsigned)(size_t)(&smem[0]);
    if (tid < 32) {
        tdm_load_2d(ldsBase + 0,    W + (size_t)tM * Cin, Cin);
        tdm_load_2d(ldsBase + 4096, xrow(0), rowlen(0));
    }
#endif

    int s = 0;
    for (int k0 = 0; k0 < Cin; k0 += 32, s ^= 1) {
#if HAVE_TDM
        if (tid < 32) {
            if (k0 + 32 < Cin) {
                const int kn = k0 + 32;
                const unsigned nb = ldsBase + (unsigned)((s ^ 1) * 8192);
                tdm_load_2d(nb + 0,    W + (size_t)tM * Cin + kn, Cin);
                tdm_load_2d(nb + 4096, xrow(kn), rowlen(kn));
                __builtin_amdgcn_s_wait_tensorcnt(2);   // current stage complete
            } else {
                __builtin_amdgcn_s_wait_tensorcnt(0);
            }
        }
        __syncthreads();
#else
        {   // fallback: manual staging (still no transpose thanks to pos-major X)
            u16* As_w = smem + s * 4096;
            u16* Bs_w = As_w + 2048;
            const u16* Xr = xrow(k0);
            const int rl = rowlen(k0);
            for (int c = tid; c < 512; c += 128) {
                int r = c >> 3, kc = (c & 7) * 4;
                *(ushort4*)&As_w[r * 32 + kc] =
                    *(const ushort4*)(W + (size_t)(tM + r) * Cin + k0 + kc);
                *(ushort4*)&Bs_w[r * 32 + kc] =
                    *(const ushort4*)(Xr + (size_t)r * rl + kc);
            }
            __syncthreads();
        }
#endif
        const u16* As = smem + s * 4096;
        const u16* Bs = As + 2048;

        v16bf afrag[2], bfrag[2];
        for (int fm = 0; fm < 2; ++fm) {
            const bf16x8* p = (const bf16x8*)&As[(wm * 32 + fm * 16 + nl) * 32];
            afrag[fm] = cat8(p[2 * half], p[2 * half + 1]);   // K = half*16 + j
        }
        for (int fn = 0; fn < 2; ++fn) {
            const bf16x8* p = (const bf16x8*)&Bs[(wn * 32 + fn * 16 + nl) * 32 + half * 16];
            bfrag[fn] = cat8(p[0], p[1]);                     // K = half*16 + j
        }
        for (int fm = 0; fm < 2; ++fm)
            for (int fn = 0; fn < 2; ++fn)
                acc[fm][fn] = wmma_bf16(afrag[fm], bfrag[fn], acc[fm][fn]);
        __syncthreads();
    }

    // ---- epilogue ----
    for (int fm = 0; fm < 2; ++fm) {
        const int rowb = tM + wm * 32 + fm * 16 + half * 8;
        for (int fn = 0; fn < 2; ++fn) {
            const int col = tN + wn * 32 + fn * 16 + nl;
            float vals[8];
            for (int r = 0; r < 8; ++r) vals[r] = acc[fm][fn][r] + bias[rowb + r];
            if (resid)
                for (int r = 0; r < 8; ++r)
                    vals[r] += resid[((size_t)b * Cout + rowb + r) * Ncol + col];
            if (Yf)
                for (int r = 0; r < 8; ++r)
                    Yf[((size_t)b * Cout + rowb + r) * Ncol + col] = vals[r];
            if (Ybf_cm)
                for (int r = 0; r < 8; ++r)
                    Ybf_cm[((size_t)b * Cout + rowb + r) * Ncol + col] = f2bf(vals[r]);
            if (Ypm) {   // pos-major: lane-contiguous 16B store
                ushort4 p0, p1;
                p0.x = f2bf(vals[0]); p0.y = f2bf(vals[1]);
                p0.z = f2bf(vals[2]); p0.w = f2bf(vals[3]);
                p1.x = f2bf(vals[4]); p1.y = f2bf(vals[5]);
                p1.z = f2bf(vals[6]); p1.w = f2bf(vals[7]);
                u16* pp = Ypm + ((size_t)b * Ncol + col) * Cout + rowb;
                *(ushort4*)pp = p0;
                *(ushort4*)(pp + 4) = p1;
            }
        }
    }
}

// =====================================================================
// Flash attention per (b, h, 64-query tile). Q/K/V channel-major bf16
// [b,d,h,n] = ((b*64+d)*8+h)*N + n. Output written POSITION-major:
// Opm[b][n][d*8+h], row length F=512.  D=64, H=8 fixed.
// =====================================================================
__global__ __launch_bounds__(128)
void flash_attn_kernel(const u16* __restrict__ Q, const u16* __restrict__ K,
                       const u16* __restrict__ V, u16* __restrict__ Opm, int Nn)
{
    __shared__ u16   Qt[64 * 64];   // [n][d]  (A frags for S)
    __shared__ u16   Kt[64 * 64];   // [m][d]  (B frags for S)
    __shared__ u16   Vt[64 * 64];   // [d][m]  (B frags for PV)
    __shared__ u16   Pt[64 * 64];   // [n][m]  (A frags for PV)
    __shared__ float Sb[64 * 64];
    __shared__ float mrow[64], lrow[64], arow[64];

    const int b = blockIdx.z, h = blockIdx.y, n0 = blockIdx.x * 64;
    const int tid = threadIdx.x, lane = tid & 31, w = tid >> 5;
    const int nl = lane & 15, half = lane >> 4;

    auto gidx = [&](int d, int n) -> size_t {
        return (((size_t)b * 64 + d) * 8 + h) * (size_t)Nn + n;
    };

    for (int c = tid; c < 1024; c += 128) {      // Q tile, transposed, pre-scaled 1/8
        int d = c >> 4, nc = (c & 15) * 4;
        ushort4 v = *(const ushort4*)(Q + gidx(d, n0 + nc));
        Qt[(nc + 0) * 64 + d] = f2bf(bf2f(v.x) * 0.125f);
        Qt[(nc + 1) * 64 + d] = f2bf(bf2f(v.y) * 0.125f);
        Qt[(nc + 2) * 64 + d] = f2bf(bf2f(v.z) * 0.125f);
        Qt[(nc + 3) * 64 + d] = f2bf(bf2f(v.w) * 0.125f);
    }
    if (tid < 64) { mrow[tid] = -1e30f; lrow[tid] = 0.f; }
    v8f oacc[4] = {};
    __syncthreads();

    for (int m0 = 0; m0 < Nn; m0 += 64) {
        for (int c = tid; c < 1024; c += 128) {
            int d = c >> 4, mc = (c & 15) * 4;
            ushort4 kv = *(const ushort4*)(K + gidx(d, m0 + mc));
            Kt[(mc + 0) * 64 + d] = kv.x;
            Kt[(mc + 1) * 64 + d] = kv.y;
            Kt[(mc + 2) * 64 + d] = kv.z;
            Kt[(mc + 3) * 64 + d] = kv.w;
            *(ushort4*)&Vt[d * 64 + mc] = *(const ushort4*)(V + gidx(d, m0 + mc));
        }
        __syncthreads();

        v8f s[4] = {};
        for (int kc = 0; kc < 64; kc += 32) {
            const bf16x8* pa = (const bf16x8*)&Qt[(w * 16 + nl) * 64 + kc];
            v16bf a = cat8(pa[2 * half], pa[2 * half + 1]);
            for (int fn = 0; fn < 4; ++fn) {
                const bf16x8* pb = (const bf16x8*)&Kt[(fn * 16 + nl) * 64 + kc + half * 16];
                s[fn] = wmma_bf16(a, cat8(pb[0], pb[1]), s[fn]);
            }
        }
        for (int fn = 0; fn < 4; ++fn)
            for (int r = 0; r < 8; ++r)
                Sb[(w * 16 + half * 8 + r) * 64 + fn * 16 + nl] = s[fn][r];
        __syncthreads();

        if (tid < 64) {                          // online softmax, one row per thread
            const float* sr = &Sb[tid * 64];
            float mo = mrow[tid], mx = mo;
            for (int j = 0; j < 64; ++j) mx = fmaxf(mx, sr[j]);
            float al = __expf(mo - mx);
            float sum = 0.f;
            u16* pr = &Pt[tid * 64];
            for (int j = 0; j < 64; ++j) {
                float e = __expf(sr[j] - mx);
                sum += e;
                pr[j] = f2bf(e);
            }
            mrow[tid] = mx;
            lrow[tid] = lrow[tid] * al + sum;
            arow[tid] = al;
        }
        __syncthreads();

        float al[8];
        for (int r = 0; r < 8; ++r) al[r] = arow[w * 16 + half * 8 + r];
        for (int fd = 0; fd < 4; ++fd)
            for (int r = 0; r < 8; ++r) oacc[fd][r] *= al[r];

        for (int kc = 0; kc < 64; kc += 32) {
            const bf16x8* pa = (const bf16x8*)&Pt[(w * 16 + nl) * 64 + kc];
            v16bf a = cat8(pa[2 * half], pa[2 * half + 1]);
            for (int fd = 0; fd < 4; ++fd) {
                const bf16x8* pb = (const bf16x8*)&Vt[(fd * 16 + nl) * 64 + kc + half * 16];
                oacc[fd] = wmma_bf16(a, cat8(pb[0], pb[1]), oacc[fd]);
            }
        }
        __syncthreads();
    }

    float inv[8];
    for (int r = 0; r < 8; ++r) inv[r] = 1.f / lrow[w * 16 + half * 8 + r];
    for (int fd = 0; fd < 4; ++fd) {
        int d = fd * 16 + nl;
        for (int r = 0; r < 8; ++r) {
            int n = n0 + w * 16 + half * 8 + r;
            Opm[((size_t)b * Nn + n) * 512 + (d * 8 + h)] = f2bf(oacc[fd][r] * inv[r]);
        }
    }
}

// =====================================================================
// Instance norm over N (biased var, eps=1e-5) + ReLU.
// Input channel-major f32 [b][C][N]; output position-major bf16 [b][N][C].
// =====================================================================
__global__ __launch_bounds__(256)
void inorm_relu_kernel(const float* __restrict__ X, u16* __restrict__ Ypm,
                       int Nn, int C)
{
    const int bb = blockIdx.x / C, c = blockIdx.x - bb * C;
    const size_t base = (size_t)blockIdx.x * Nn;
    __shared__ float s1[256], s2[256];
    float a = 0.f, q = 0.f;
    for (int i = threadIdx.x; i < Nn; i += 256) {
        float x = X[base + i];
        a += x; q += x * x;
    }
    s1[threadIdx.x] = a; s2[threadIdx.x] = q;
    __syncthreads();
    for (int s = 128; s > 0; s >>= 1) {
        if ((int)threadIdx.x < s) {
            s1[threadIdx.x] += s1[threadIdx.x + s];
            s2[threadIdx.x] += s2[threadIdx.x + s];
        }
        __syncthreads();
    }
    float mean = s1[0] / (float)Nn;
    float var  = s2[0] / (float)Nn - mean * mean;
    float rs   = rsqrtf(var + 1e-5f);
    for (int i = threadIdx.x; i < Nn; i += 256) {
        float v = (X[base + i] - mean) * rs;
        Ypm[((size_t)bb * Nn + i) * C + c] = f2bf(fmaxf(v, 0.f));
    }
}

// =====================================================================
// host side
// =====================================================================
extern "C" void kernel_launch(void* const* d_in, const int* in_sizes, int n_in,
                              void* d_out, int out_size, void* d_ws, size_t ws_size,
                              hipStream_t stream)
{
    (void)in_sizes; (void)n_in; (void)out_size; (void)ws_size;
    const int B = 2, F = 512, N = 2048, C2 = 1024, H = 8;
    const size_t BFN  = (size_t)B * F * N;
    const size_t BC2N = (size_t)B * C2 * N;

    const float* src = (const float*)d_in[0];
    const float* tgt = (const float*)d_in[1];
    const float* Wq = (const float*)d_in[2];  const float* bq = (const float*)d_in[3];
    const float* Wk = (const float*)d_in[4];  const float* bk = (const float*)d_in[5];
    const float* Wv = (const float*)d_in[6];  const float* bv = (const float*)d_in[7];
    const float* Wm = (const float*)d_in[8];  const float* bm = (const float*)d_in[9];
    const float* W1 = (const float*)d_in[10]; const float* b1 = (const float*)d_in[11];
    const float* W2 = (const float*)d_in[12]; const float* b2 = (const float*)d_in[13];
    float* out = (float*)d_out;

    char* ws = (char*)d_ws;
    size_t off = 0;
    auto alloc = [&](size_t bytes) -> void* {
        void* p = ws + off;
        off += (bytes + 255) & ~(size_t)255;
        return p;
    };
    u16* wq_bf = (u16*)alloc((size_t)F * F * 2);
    u16* wk_bf = (u16*)alloc((size_t)F * F * 2);
    u16* wv_bf = (u16*)alloc((size_t)F * F * 2);
    u16* wm_bf = (u16*)alloc((size_t)F * F * 2);
    u16* w1_bf = (u16*)alloc((size_t)C2 * C2 * 2);
    u16* w2_bf = (u16*)alloc((size_t)F * C2 * 2);
    u16* src_pm    = (u16*)alloc(BFN * 2);   // [b][n][c] bf16
    u16* tgt_pm    = (u16*)alloc(BFN * 2);
    u16* srcout_pm = (u16*)alloc(BFN * 2);
    u16* q_cm    = (u16*)alloc(BFN * 2);     // [b][f][n] bf16 (attention layout)
    u16* k_cm    = (u16*)alloc(BFN * 2);
    u16* v_cm    = (u16*)alloc(BFN * 2);
    u16* attn_pm = (u16*)alloc(BFN * 2);
    u16* a_pm    = (u16*)alloc(BFN * 2);
    float* t_f32 = (float*)alloc(BC2N * 4);  // channel-major (for instance norm)
    u16* h_pm    = (u16*)alloc(BC2N * 2);

    auto cvt = [&](const float* x, u16* y, size_t n) {
        f32_to_bf16_kernel<<<dim3((unsigned)((n + 255) / 256)), 256, 0, stream>>>(x, y, (int)n);
    };
    cvt(Wq, wq_bf, (size_t)F * F);
    cvt(Wk, wk_bf, (size_t)F * F);
    cvt(Wv, wv_bf, (size_t)F * F);
    cvt(Wm, wm_bf, (size_t)F * F);
    cvt(W1, w1_bf, (size_t)C2 * C2);
    cvt(W2, w2_bf, (size_t)F * C2);
    {
        dim3 g((unsigned)(((size_t)F * N + 255) / 256), 1, B);
        cm_to_pm_bf16_kernel<<<g, 256, 0, stream>>>(src, src_pm, F, N);
        cm_to_pm_bf16_kernel<<<g, 256, 0, stream>>>(tgt, tgt_pm, F, N);
    }

    auto gemm = [&](const u16* W, const u16* X0, const u16* X1, int K0,
                    const float* bias, const float* resid,
                    float* Yf, u16* Ybf_cm, u16* Ypm, int Cout, int Cin) {
        dim3 g(N / 64, Cout / 64, B);
        gemm_bf16_kernel<<<g, 128, 0, stream>>>(W, X0, X1, K0, bias, resid,
                                                Yf, Ybf_cm, Ypm, Cout, Cin, N);
    };

    auto run_block = [&](const float* x_f32, const u16* x_pm, const u16* other_pm,
                         float* out_f32, u16* out_pm) {
        gemm(wq_bf, x_pm,     x_pm,     F, bq, nullptr, nullptr, q_cm, nullptr, F, F);
        gemm(wk_bf, other_pm, other_pm, F, bk, nullptr, nullptr, k_cm, nullptr, F, F);
        gemm(wv_bf, other_pm, other_pm, F, bv, nullptr, nullptr, v_cm, nullptr, F, F);
        flash_attn_kernel<<<dim3(N / 64, H, B), 128, 0, stream>>>(q_cm, k_cm, v_cm, attn_pm, N);
        gemm(wm_bf, attn_pm, attn_pm, F, bm, nullptr, nullptr, nullptr, a_pm, F, F);
        // W1 @ concat(x, a): virtual K-concat of two position-major buffers
        gemm(w1_bf, x_pm, a_pm, F, b1, nullptr, t_f32, nullptr, nullptr, C2, C2);
        inorm_relu_kernel<<<dim3(B * C2), 256, 0, stream>>>(t_f32, h_pm, N, C2);
        // out = x + W2 @ h   (f32 to d_out; bf16 pos-major copy for next block)
        gemm(w2_bf, h_pm, h_pm, C2, b2, x_f32, out_f32, nullptr, out_pm, F, C2);
    };

    run_block(src, src_pm, tgt_pm, out, srcout_pm);            // src_out
    run_block(tgt, tgt_pm, srcout_pm, out + BFN, nullptr);     // tgt_out (attends updated src)
}
